// MaskedTransformer_13846974562415
// MI455X (gfx1250) — compile-verified, matrix operand order checked by reading, not run on hardware
//
#include <hip/hip_runtime.h>

// ---------------------------------------------------------------------------
// Shapes (fixed by the reference)
// ---------------------------------------------------------------------------
#define BB   32
#define TT   256
#define NE   768
#define NH   12
#define DH   64
#define NL   8
#define FF   3072          // 4*NE
#define CONV_K 18432       // 2048*9
#define ROWS (BB*TT)       // 8192

typedef __attribute__((ext_vector_type(8)))  float        v8f;
typedef __attribute__((ext_vector_type(8)))  unsigned int v8u;
typedef __attribute__((ext_vector_type(4)))  unsigned int u32x4;
typedef __attribute__((ext_vector_type(4)))  int          i32x4;
typedef __attribute__((ext_vector_type(16))) __bf16       v16bf;

#define AS1 __attribute__((address_space(1)))
#define AS3 __attribute__((address_space(3)))

// Async global->LDS (CDNA5 ASYNCcnt path) with sync fallback
#if __has_builtin(__builtin_amdgcn_global_load_async_to_lds_b128) && \
    __has_builtin(__builtin_amdgcn_s_wait_asynccnt)
#define HAVE_ASYNC_LDS 1
__device__ __forceinline__ void cp16_g2l(const unsigned short* g, unsigned short* l) {
    __builtin_amdgcn_global_load_async_to_lds_b128((AS1 i32x4*)g, (AS3 i32x4*)l, 0, 0);
}
__device__ __forceinline__ void cp_wait() { __builtin_amdgcn_s_wait_asynccnt(0); }
#else
#define HAVE_ASYNC_LDS 0
__device__ __forceinline__ void cp16_g2l(const unsigned short* g, unsigned short* l) {
    *(u32x4*)l = *(const u32x4*)g;
}
__device__ __forceinline__ void cp_wait() {}
#endif

__device__ __forceinline__ unsigned short f2bf(float f) {
    unsigned u = __builtin_bit_cast(unsigned, f);
    u += 0x7fffu + ((u >> 16) & 1u);
    return (unsigned short)(u >> 16);
}
__device__ __forceinline__ float geluf(float v) {
    return 0.5f * v * (1.0f + erff(v * 0.70710678118654752f));
}

// ---------------------------------------------------------------------------
// fp32 -> bf16 elementwise convert
// ---------------------------------------------------------------------------
__global__ __launch_bounds__(256) void cvt_bf16_kernel(const float* __restrict__ in,
                                                       unsigned short* __restrict__ out,
                                                       long n) {
    long i = (long)blockIdx.x * 256 + threadIdx.x;
    long stride = (long)gridDim.x * 256;
    for (; i < n; i += stride) out[i] = f2bf(in[i]);
}

// conv_w [384,2048,3,3] fp32 -> Bmat [18432,384] bf16  (k = c*9+ky*3+kx, n = oc)
__global__ __launch_bounds__(256) void repack_convw_kernel(const float* __restrict__ w,
                                                           unsigned short* __restrict__ out) {
    long n = 384L * CONV_K;
    long i = (long)blockIdx.x * 256 + threadIdx.x;
    long stride = (long)gridDim.x * 256;
    for (; i < n; i += stride) {
        long oc  = i / CONV_K;
        long rem = i - oc * CONV_K;      // c*9 + ky*3 + kx
        out[rem * 384 + oc] = f2bf(w[i]);
    }
}

// ---------------------------------------------------------------------------
// LayerNorm: fp32 [rows, 768] in  ->  bf16 [rows, 768] out
// ---------------------------------------------------------------------------
__global__ __launch_bounds__(256) void ln_kernel(const float* __restrict__ x,
                                                 const float* __restrict__ g,
                                                 const float* __restrict__ b,
                                                 unsigned short* __restrict__ out) {
    __shared__ float s1[256];
    __shared__ float s2[256];
    const int tid = threadIdx.x;
    const long row = blockIdx.x;
    const float* xr = x + row * NE;
    float a0 = xr[tid], a1 = xr[tid + 256], a2 = xr[tid + 512];
    s1[tid] = a0 + a1 + a2;
    s2[tid] = a0 * a0 + a1 * a1 + a2 * a2;
    __syncthreads();
    for (int o = 128; o > 0; o >>= 1) {
        if (tid < o) { s1[tid] += s1[tid + o]; s2[tid] += s2[tid + o]; }
        __syncthreads();
    }
    const float m  = s1[0] * (1.0f / NE);
    const float rs = rsqrtf(s2[0] * (1.0f / NE) - m * m + 1e-5f);
    out[row * NE + tid      ] = f2bf((a0 - m) * rs * g[tid      ] + b[tid      ]);
    out[row * NE + tid + 256] = f2bf((a1 - m) * rs * g[tid + 256] + b[tid + 256]);
    out[row * NE + tid + 512] = f2bf((a2 - m) * rs * g[tid + 512] + b[tid + 512]);
}

// ---------------------------------------------------------------------------
// vm/fm projections + positional embedding (tiny FLOPs, plain VALU)
// ---------------------------------------------------------------------------
__global__ __launch_bounds__(256) void embed_kernel(float* __restrict__ x,
                                                    const float* __restrict__ vmp,
                                                    const float* __restrict__ fmp,
                                                    const float* __restrict__ vm_w,
                                                    const float* __restrict__ vm_b,
                                                    const float* __restrict__ fm_w,
                                                    const float* __restrict__ fm_b,
                                                    const float* __restrict__ pos) {
    const long row = blockIdx.x;
    const int  t   = (int)(row & 255);
    for (int c = threadIdx.x; c < NE; c += 256) {
        float pv = pos[(long)t * NE + c];
        if (c < 384) {
            x[row * NE + c] += pv;                 // conv part already there
        } else if (c < 576) {
            int j = c - 384; float s = 0.f;
            const float* vr = vmp + row * 256;
            for (int i = 0; i < 256; ++i) s += vr[i] * vm_w[i * 192 + j];
            x[row * NE + c] = s + vm_b[j] + pv;
        } else {
            int j = c - 576; float s = 0.f;
            const float* fr = fmp + row * 256;
            for (int i = 0; i < 256; ++i) s += fr[i] * fm_w[i * 192 + j];
            x[row * NE + c] = s + fm_b[j] + pv;
        }
    }
}

// ---------------------------------------------------------------------------
// Generic bf16 WMMA GEMM:  C[M,N] = A[M,K] * B[K,N] (+bias, +gelu, epilogues)
//   block = 256 threads = 8 waves; tile 128x128, BK=32 (one WMMA k-step)
//   wave grid 4x2 -> each wave a 32x64 tile = 2x4 WMMA 16x16 accumulators
//   Double-buffered LDS staging via GLOBAL_LOAD_ASYNC_TO_LDS_B128 (ASYNCcnt),
//   overlapping HBM/L2 traffic with WMMA compute.
// ACONV:   0 = A is bf16 row-major [M,K] (lda)   1 = A is bf16 img [B,2048,16,16], im2col gather
// OUTMODE: 0 = bf16 store  1 = f32 store  2 = f32 residual add  3 = bf16 QKV head-scatter
// ---------------------------------------------------------------------------
#define BM 128
#define BN 128
#define BK 32
#define LDA_S 40          // bf16 elems per A row in LDS  (row pitch 80 B, 16B-aligned)
#define LDB_S 136         // bf16 elems per B row in LDS  (row pitch 272 B, 16B-aligned)

template <int ACONV, int OUTMODE, int DOGELU>
__global__ __launch_bounds__(256) void gemm_bf16_kernel(
    const unsigned short* __restrict__ A, int lda,
    const unsigned short* __restrict__ Bw, int N,
    const float* __restrict__ bias,
    float* __restrict__ Cf, unsigned short* __restrict__ Cb,
    int ldc, int cofs, int K) {

    __shared__ unsigned short lA[2][BM * LDA_S];
    __shared__ unsigned short lB[2][BK * LDB_S];   // untransposed: [k][n]

    const int tid  = threadIdx.x;
    const int lane = tid & 31;
    const int w    = tid >> 5;
    const int m0   = blockIdx.y * BM;
    const int n0   = blockIdx.x * BN;
    const int mw   = (w >> 1) * 32;
    const int nw   = (w & 1) * 64;

    v8f acc[2][4];
#pragma unroll
    for (int i = 0; i < 2; ++i)
#pragma unroll
        for (int j = 0; j < 4; ++j)
#pragma unroll
            for (int e = 0; e < 8; ++e) acc[i][j][e] = 0.0f;

    auto stage = [&](int k0, int buf) {
        if (ACONV) {
            // on-the-fly 3x3 im2col gather (zero-padded borders), scalar
#pragma unroll
            for (int it = 0; it < 16; ++it) {
                int i  = tid + it * 256;
                int mm = i >> 5, kk = i & 31;
                int r  = m0 + mm, kg = k0 + kk;
                int bb = r >> 8, sp = r & 255, yy = sp >> 4, xx = sp & 15;
                int c  = kg / 9, rem = kg - c * 9, ky = rem / 3, kx = rem - ky * 3;
                int iy = yy + ky - 1, ix = xx + kx - 1;
                unsigned short val = 0;
                if ((unsigned)iy < 16u && (unsigned)ix < 16u)
                    val = A[((((long)bb << 11) + c) << 8) + (iy << 4) + ix];
                lA[buf][mm * LDA_S + kk] = val;
            }
        } else {
            // A tile: 128x32 bf16, one b128 per lane x 2 passes
#pragma unroll
            for (int it = 0; it < 2; ++it) {
                int i  = tid + it * 256;
                int mm = i >> 2, kb = (i & 3) * 8;
                cp16_g2l(A + (long)(m0 + mm) * lda + k0 + kb,
                         &lA[buf][mm * LDA_S + kb]);
            }
        }
        // B tile: 32x128 bf16, one b128 per lane x 2 passes
#pragma unroll
        for (int it = 0; it < 2; ++it) {
            int i  = tid + it * 256;
            int kk = i >> 4, nb = (i & 15) * 8;
            cp16_g2l(Bw + (long)(k0 + kk) * N + n0 + nb,
                     &lB[buf][kk * LDB_S + nb]);
        }
    };

    stage(0, 0);
    cp_wait();
    __syncthreads();

    const int nk = K / BK;
    for (int t = 0; t < nk; ++t) {
        const int cur = t & 1;
        if (t + 1 < nk) stage((t + 1) * BK, cur ^ 1);   // overlap with compute below

        // ---- fragments from lA[cur]/lB[cur] (documented 16-bit WMMA layout) ----
        v16bf af[2], bfr[4];
#pragma unroll
        for (int mf = 0; mf < 2; ++mf) {
            int m = mw + mf * 16 + (lane & 15);
            v8u u;
#pragma unroll
            for (int p = 0; p < 8; ++p) {
                int kk = ((p & 4) ? 16 : 0) + ((lane & 16) ? 8 : 0) + ((p & 3) << 1);
                u[p] = *(const unsigned*)&lA[cur][m * LDA_S + kk];
            }
            af[mf] = __builtin_bit_cast(v16bf, u);
        }
#pragma unroll
        for (int nf = 0; nf < 4; ++nf) {
            int n = nw + nf * 16 + (lane & 15);
            v8u u;
#pragma unroll
            for (int p = 0; p < 8; ++p) {
                int kk = ((p & 4) ? 16 : 0) + ((lane & 16) ? 8 : 0) + ((p & 3) << 1);
                unsigned lo = lB[cur][(kk    ) * LDB_S + n];
                unsigned hi = lB[cur][(kk + 1) * LDB_S + n];
                u[p] = lo | (hi << 16);
            }
            bfr[nf] = __builtin_bit_cast(v16bf, u);
        }
#pragma unroll
        for (int mf = 0; mf < 2; ++mf)
#pragma unroll
            for (int nf = 0; nf < 4; ++nf)
                acc[mf][nf] = __builtin_amdgcn_wmma_f32_16x16x32_bf16(
                    false, af[mf], false, bfr[nf], (short)0, acc[mf][nf], false, false);

        cp_wait();          // next tile landed in LDS
        __syncthreads();
    }

    // ---- epilogue ----
#pragma unroll
    for (int mf = 0; mf < 2; ++mf)
#pragma unroll
        for (int nf = 0; nf < 4; ++nf)
#pragma unroll
            for (int r = 0; r < 8; ++r) {
                int row = m0 + mw + mf * 16 + r + ((lane >> 4) << 3);
                int col = n0 + nw + nf * 16 + (lane & 15);
                float v = acc[mf][nf][r];
                if (bias) v += bias[col];
                if (DOGELU) v = geluf(v);
                if (OUTMODE == 0) {
                    Cb[(long)row * ldc + cofs + col] = f2bf(v);
                } else if (OUTMODE == 1) {
                    Cf[(long)row * ldc + cofs + col] = v;
                } else if (OUTMODE == 2) {
                    long o = (long)row * ldc + cofs + col;
                    Cf[o] += v;
                } else {  // QKV scatter: [M=b*T+t, N=h*DH+d] -> [b,h,t,d]
                    int hh = col >> 6, d = col & 63, bb = row >> 8, t = row & 255;
                    Cb[((((long)bb * NH + hh) << 8) + t) * DH + d] = f2bf(v);
                }
            }
}

// ---------------------------------------------------------------------------
// Attention: one workgroup per (b*NH + h, 64-row block).
// Q/K/V bf16 [B*NH, T, DH]; score block 64x256 fp32 in LDS; O -> Y [B,T,NE] bf16.
// Entire K/V/score block resident in CDNA5's 320KB WGP LDS.
// ---------------------------------------------------------------------------
#define QLD 72
#define KLD 72
#define VLD 72
#define SLD 260
#define PLD 264
#define ATTN_SMEM ((64*QLD + 256*KLD + 256*VLD + 64*PLD) * 2 + (64*SLD) * 4 + 512 * 4)

__global__ __launch_bounds__(256) void attn_kernel(const unsigned short* __restrict__ Q,
                                                   const unsigned short* __restrict__ Km,
                                                   const unsigned short* __restrict__ V,
                                                   unsigned short* __restrict__ Y,
                                                   float scale) {
    extern __shared__ char smem[];
    unsigned short* lq   = (unsigned short*)smem;                  // 64 x QLD
    unsigned short* lk   = lq + 64 * QLD;                          // 256 x KLD
    unsigned short* lv   = lk + 256 * KLD;                         // 256 x VLD
    float*          lss  = (float*)(lv + 256 * VLD);               // 64 x SLD fp32
    unsigned short* lp   = (unsigned short*)(lss + 64 * SLD);      // 64 x PLD bf16
    float*          pmax = (float*)(lp + 64 * PLD);                // 256
    float*          psum = pmax + 256;                             // 256

    const int tid  = threadIdx.x;
    const int lane = tid & 31;
    const int w    = tid >> 5;
    const int bh   = blockIdx.x >> 2;
    const int m0   = (blockIdx.x & 3) * 64;

    const unsigned short* qb = Q  + (long)bh * TT * DH;
    const unsigned short* kb = Km + (long)bh * TT * DH;
    const unsigned short* vb = V  + (long)bh * TT * DH;

    // async-stage Q (64x64) and K/V (256x64) as b128 per lane
    for (int i = tid; i < 64 * 8; i += 256) {
        int r = i >> 3, cb = (i & 7) * 8;
        cp16_g2l(qb + (long)(m0 + r) * DH + cb, &lq[r * QLD + cb]);
    }
    for (int i = tid; i < 256 * 8; i += 256) {
        int r = i >> 3, cb = (i & 7) * 8;
        cp16_g2l(kb + (long)r * DH + cb, &lk[r * KLD + cb]);
        cp16_g2l(vb + (long)r * DH + cb, &lv[r * VLD + cb]);
    }
    cp_wait();
    __syncthreads();

    // ---- S = scale * Q @ K^T : per wave 64 rows x 32 cols (4x2 frags) ----
    {
        v8f acc[4][2];
#pragma unroll
        for (int i = 0; i < 4; ++i)
#pragma unroll
            for (int j = 0; j < 2; ++j)
#pragma unroll
                for (int e = 0; e < 8; ++e) acc[i][j][e] = 0.0f;

        for (int k0 = 0; k0 < DH; k0 += 32) {
            v16bf aq[4], bk2[2];
#pragma unroll
            for (int mf = 0; mf < 4; ++mf) {
                int m = mf * 16 + (lane & 15);
                v8u u;
#pragma unroll
                for (int p = 0; p < 8; ++p) {
                    int kk = ((p & 4) ? 16 : 0) + ((lane & 16) ? 8 : 0) + ((p & 3) << 1);
                    u[p] = *(const unsigned*)&lq[m * QLD + k0 + kk];
                }
                aq[mf] = __builtin_bit_cast(v16bf, u);
            }
#pragma unroll
            for (int nf = 0; nf < 2; ++nf) {
                int n = w * 32 + nf * 16 + (lane & 15);   // key index
                v8u u;
#pragma unroll
                for (int p = 0; p < 8; ++p) {
                    int kk = ((p & 4) ? 16 : 0) + ((lane & 16) ? 8 : 0) + ((p & 3) << 1);
                    u[p] = *(const unsigned*)&lk[n * KLD + k0 + kk];  // K^T pairs along dh
                }
                bk2[nf] = __builtin_bit_cast(v16bf, u);
            }
#pragma unroll
            for (int mf = 0; mf < 4; ++mf)
#pragma unroll
                for (int nf = 0; nf < 2; ++nf)
                    acc[mf][nf] = __builtin_amdgcn_wmma_f32_16x16x32_bf16(
                        false, aq[mf], false, bk2[nf], (short)0, acc[mf][nf], false, false);
        }
#pragma unroll
        for (int mf = 0; mf < 4; ++mf)
#pragma unroll
            for (int nf = 0; nf < 2; ++nf)
#pragma unroll
                for (int r = 0; r < 8; ++r) {
                    int row = mf * 16 + r + ((lane >> 4) << 3);
                    int col = w * 32 + nf * 16 + (lane & 15);
                    lss[row * SLD + col] = acc[mf][nf][r] * scale;
                }
    }
    __syncthreads();

    // ---- softmax over 256 cols; 4 threads per row ----
    {
        int row = tid >> 2, seg = tid & 3;
        float* sr = lss + row * SLD;
        float mx = -3.0e38f;
        for (int c = seg * 64; c < seg * 64 + 64; ++c) mx = fmaxf(mx, sr[c]);
        pmax[row * 4 + seg] = mx;
        __syncthreads();
        float rm = fmaxf(fmaxf(pmax[row * 4 + 0], pmax[row * 4 + 1]),
                         fmaxf(pmax[row * 4 + 2], pmax[row * 4 + 3]));
        float s = 0.f;
        for (int c = seg * 64; c < seg * 64 + 64; ++c) {
            float e = __expf(sr[c] - rm);
            sr[c] = e; s += e;
        }
        psum[row * 4 + seg] = s;
        __syncthreads();
        float inv = 1.0f / (psum[row * 4 + 0] + psum[row * 4 + 1] +
                            psum[row * 4 + 2] + psum[row * 4 + 3]);
        for (int c = seg * 64; c < seg * 64 + 64; ++c)
            lp[row * PLD + c] = f2bf(sr[c] * inv);
    }
    __syncthreads();

    // ---- O = P @ V : wave grid 4x2, each wave 16 rows x 32 cols ----
    {
        const int wr = w >> 1, wc = w & 1;
        v8f acc[2];
#pragma unroll
        for (int j = 0; j < 2; ++j)
#pragma unroll
            for (int e = 0; e < 8; ++e) acc[j][e] = 0.0f;

        for (int k0 = 0; k0 < 256; k0 += 32) {
            v16bf ap, bv2[2];
            {
                int m = wr * 16 + (lane & 15);
                v8u u;
#pragma unroll
                for (int p = 0; p < 8; ++p) {
                    int kk = ((p & 4) ? 16 : 0) + ((lane & 16) ? 8 : 0) + ((p & 3) << 1);
                    u[p] = *(const unsigned*)&lp[m * PLD + k0 + kk];
                }
                ap = __builtin_bit_cast(v16bf, u);
            }
#pragma unroll
            for (int nf = 0; nf < 2; ++nf) {
                int n = wc * 32 + nf * 16 + (lane & 15);
                v8u u;
#pragma unroll
                for (int p = 0; p < 8; ++p) {
                    int kk = ((p & 4) ? 16 : 0) + ((lane & 16) ? 8 : 0) + ((p & 3) << 1);
                    unsigned lo = lv[(k0 + kk    ) * VLD + n];
                    unsigned hi = lv[(k0 + kk + 1) * VLD + n];
                    u[p] = lo | (hi << 16);
                }
                bv2[nf] = __builtin_bit_cast(v16bf, u);
            }
#pragma unroll
            for (int nf = 0; nf < 2; ++nf)
                acc[nf] = __builtin_amdgcn_wmma_f32_16x16x32_bf16(
                    false, ap, false, bv2[nf], (short)0, acc[nf], false, false);
        }
        const int b = bh / NH, h = bh - b * NH;
#pragma unroll
        for (int nf = 0; nf < 2; ++nf)
#pragma unroll
            for (int r = 0; r < 8; ++r) {
                int row = m0 + wr * 16 + r + ((lane >> 4) << 3);
                int col = wc * 32 + nf * 16 + (lane & 15);
                Y[((long)(b * TT + row)) * NE + h * DH + col] = f2bf(acc[nf][r]);
            }
    }
}

// ---------------------------------------------------------------------------
// Host orchestration
// ---------------------------------------------------------------------------
extern "C" void kernel_launch(void* const* d_in, const int* in_sizes, int n_in,
                              void* d_out, int out_size, void* d_ws, size_t ws_size,
                              hipStream_t stream) {
    (void)in_sizes; (void)n_in; (void)out_size; (void)ws_size;

    const float* img      = (const float*)d_in[0];
    const float* vmp      = (const float*)d_in[1];
    const float* fmp      = (const float*)d_in[2];
    const float* conv_w   = (const float*)d_in[3];
    const float* conv_b   = (const float*)d_in[4];
    const float* vm_w     = (const float*)d_in[5];
    const float* vm_b     = (const float*)d_in[6];
    const float* fm_w     = (const float*)d_in[7];
    const float* fm_b     = (const float*)d_in[8];
    const float* pos      = (const float*)d_in[9];
    const float* ln1_g    = (const float*)d_in[10];
    const float* ln1_b    = (const float*)d_in[11];
    const float* Wq       = (const float*)d_in[12];
    const float* bq       = (const float*)d_in[13];
    const float* Wk       = (const float*)d_in[14];
    const float* bk       = (const float*)d_in[15];
    const float* Wv       = (const float*)d_in[16];
    const float* bv       = (const float*)d_in[17];
    const float* Wo       = (const float*)d_in[18];
    const float* bo       = (const float*)d_in[19];
    const float* ln2_g    = (const float*)d_in[20];
    const float* ln2_b    = (const float*)d_in[21];
    const float* W1       = (const float*)d_in[22];
    const float* b1       = (const float*)d_in[23];
    const float* W2       = (const float*)d_in[24];
    const float* b2       = (const float*)d_in[25];
    const float* dln1_g   = (const float*)d_in[26];
    const float* dln1_b   = (const float*)d_in[27];
    const float* dec_w    = (const float*)d_in[28];
    const float* dec_b    = (const float*)d_in[29];
    const float* dln2_g   = (const float*)d_in[30];
    const float* dln2_b   = (const float*)d_in[31];
    const float* head_w   = (const float*)d_in[32];
    const float* head_b   = (const float*)d_in[33];
    float* out = (float*)d_out;

    // ---- workspace layout ----
    char* ws = (char*)d_ws;
    size_t off = 0;
    auto take = [&](size_t bytes) -> char* {
        char* p = ws + off;
        off += (bytes + 255) & ~(size_t)255;
        return p;
    };
    float*          X    = (float*)take((size_t)ROWS * NE * 4);
    unsigned short* Hb   = (unsigned short*)take((size_t)ROWS * NE * 2);
    unsigned short* Qb   = (unsigned short*)take((size_t)ROWS * NE * 2);
    unsigned short* Kb   = (unsigned short*)take((size_t)ROWS * NE * 2);
    unsigned short* Vb   = (unsigned short*)take((size_t)ROWS * NE * 2);
    unsigned short* Yb   = (unsigned short*)take((size_t)ROWS * NE * 2);
    char*           MLPr = take((size_t)ROWS * FF * 2);
    unsigned short* MLPb   = (unsigned short*)MLPr;
    unsigned short* IMGb   = (unsigned short*)MLPr;                  // alias (embed phase)
    unsigned short* CONVWb = (unsigned short*)(MLPr + 33554432);     // alias (embed phase)
    float*          XT     = (float*)MLPr;                           // alias (head phase)
    unsigned short* WQb  = (unsigned short*)take((size_t)NL * NE * NE * 2);
    unsigned short* WKb  = (unsigned short*)take((size_t)NL * NE * NE * 2);
    unsigned short* WVb  = (unsigned short*)take((size_t)NL * NE * NE * 2);
    unsigned short* WOb  = (unsigned short*)take((size_t)NL * NE * NE * 2);
    unsigned short* W1b  = (unsigned short*)take((size_t)NL * NE * FF * 2);
    unsigned short* W2b  = (unsigned short*)take((size_t)NL * NE * FF * 2);
    unsigned short* DECb = (unsigned short*)take((size_t)NE * NE * 2);
    unsigned short* HWb  = (unsigned short*)take((size_t)NE * 256 * 2);

    const long nWqkv = (long)NL * NE * NE;
    const long nWff  = (long)NL * NE * FF;

    // ---- convert weights & image to bf16 ----
    cvt_bf16_kernel<<<4096, 256, 0, stream>>>(img,    IMGb, 32L * 2048 * 256);
    repack_convw_kernel<<<4096, 256, 0, stream>>>(conv_w, CONVWb);
    cvt_bf16_kernel<<<4096, 256, 0, stream>>>(Wq,     WQb,  nWqkv);
    cvt_bf16_kernel<<<4096, 256, 0, stream>>>(Wk,     WKb,  nWqkv);
    cvt_bf16_kernel<<<4096, 256, 0, stream>>>(Wv,     WVb,  nWqkv);
    cvt_bf16_kernel<<<4096, 256, 0, stream>>>(Wo,     WOb,  nWqkv);
    cvt_bf16_kernel<<<4096, 256, 0, stream>>>(W1,     W1b,  nWff);
    cvt_bf16_kernel<<<4096, 256, 0, stream>>>(W2,     W2b,  nWff);
    cvt_bf16_kernel<<<1024, 256, 0, stream>>>(dec_w,  DECb, (long)NE * NE);
    cvt_bf16_kernel<<<1024, 256, 0, stream>>>(head_w, HWb,  (long)NE * 256);

    // ---- token embedding ----
    // conv implicit GEMM: M=8192, N=384, K=18432 -> X[:, 0:384] fp32 (+conv_b)
    gemm_bf16_kernel<1, 1, 0><<<dim3(3, 64), 256, 0, stream>>>(
        IMGb, 0, CONVWb, 384, conv_b, X, nullptr, NE, 0, CONV_K);
    embed_kernel<<<ROWS, 256, 0, stream>>>(X, vmp, fmp, vm_w, vm_b, fm_w, fm_b, pos);

    // ---- transformer layers ----
    const float scale = 0.125f;  // 1/sqrt(64)
    for (int L = 0; L < NL; ++L) {
        const long wofs  = (long)L * NE * NE;
        const long wofsF = (long)L * NE * FF;
        ln_kernel<<<ROWS, 256, 0, stream>>>(X, ln1_g + L * NE, ln1_b + L * NE, Hb);
        gemm_bf16_kernel<0, 3, 0><<<dim3(6, 64), 256, 0, stream>>>(
            Hb, NE, WQb + wofs, NE, bq + L * NE, nullptr, Qb, 0, 0, NE);
        gemm_bf16_kernel<0, 3, 0><<<dim3(6, 64), 256, 0, stream>>>(
            Hb, NE, WKb + wofs, NE, bk + L * NE, nullptr, Kb, 0, 0, NE);
        gemm_bf16_kernel<0, 3, 0><<<dim3(6, 64), 256, 0, stream>>>(
            Hb, NE, WVb + wofs, NE, bv + L * NE, nullptr, Vb, 0, 0, NE);
        attn_kernel<<<BB * NH * 4, 256, ATTN_SMEM, stream>>>(Qb, Kb, Vb, Yb, scale);
        gemm_bf16_kernel<0, 2, 0><<<dim3(6, 64), 256, 0, stream>>>(
            Yb, NE, WOb + wofs, NE, bo + L * NE, X, nullptr, NE, 0, NE);
        ln_kernel<<<ROWS, 256, 0, stream>>>(X, ln2_g + L * NE, ln2_b + L * NE, Hb);
        gemm_bf16_kernel<0, 0, 1><<<dim3(24, 64), 256, 0, stream>>>(
            Hb, NE, W1b + wofsF, FF, b1 + L * FF, nullptr, MLPb, FF, 0, NE);
        gemm_bf16_kernel<0, 2, 0><<<dim3(6, 64), 256, 0, stream>>>(
            MLPb, FF, W2b + wofsF, NE, b2 + L * NE, X, nullptr, NE, 0, FF);
    }

    // ---- prediction head ----
    ln_kernel<<<ROWS, 256, 0, stream>>>(X, dln1_g, dln1_b, Hb);
    gemm_bf16_kernel<0, 1, 1><<<dim3(6, 64), 256, 0, stream>>>(
        Hb, NE, DECb, NE, dec_b, XT, nullptr, NE, 0, NE);
    ln_kernel<<<ROWS, 256, 0, stream>>>(XT, dln2_g, dln2_b, Hb);
    gemm_bf16_kernel<0, 1, 0><<<dim3(2, 64), 256, 0, stream>>>(
        Hb, NE, HWb, 256, head_b, out, nullptr, 256, 0, NE);
}
